// MultiHeadSelfAttention_71365176590902
// MI455X (gfx1250) — compile-verified, matrix operand order checked by reading, not run on hardware
//
#include <hip/hip_runtime.h>

#define DEV __device__ __forceinline__

typedef __attribute__((ext_vector_type(16))) __bf16          v16bf;
typedef __attribute__((ext_vector_type(8)))  float           v8f;
typedef __attribute__((ext_vector_type(8)))  unsigned short  us8;
typedef __attribute__((ext_vector_type(4)))  unsigned short  us4;

static constexpr int Bb   = 2;
static constexpr int Ss   = 4096;
static constexpr int Dd   = 512;
static constexpr int Hh   = 8;
static constexpr int DK   = 64;
static constexpr int ROWS = Bb * Ss;  // 8192

// ---------- helpers ----------

DEV unsigned short f2bf(float f) {
  unsigned u = __float_as_uint(f);
  u += 0x7fffu + ((u >> 16) & 1u);   // round-to-nearest-even
  return (unsigned short)(u >> 16);
}

union FragU { v16bf v; us8 h[2]; };

// Load a 16x32 bf16 fragment (A layout, per ISA 7.12.2) from a row-major
// [*, ld] bf16 matrix starting at (row, k0). Works for global or LDS
// pointers (inlined; address space inferred).
DEV v16bf load_frag(const unsigned short* base, int row, int k0, int ld,
                    int lane) {
  const int m = lane & 15, hi = lane >> 4;
  const unsigned short* p =
      base + (size_t)(row + m) * (size_t)ld + (size_t)(k0 + 8 * hi);
  FragU f;
  f.h[0] = *(const us8*)(p);        // K = 8*hi .. 8*hi+7
  f.h[1] = *(const us8*)(p + 16);   // K = 16+8*hi .. 16+8*hi+7
  return f.v;
}

DEV v8f wmma_bf16(v16bf a, v16bf b, v8f c) {
  return __builtin_amdgcn_wmma_f32_16x16x32_bf16(
      /*neg_a=*/false, a, /*neg_b=*/false, b,
      /*c_mod=*/(short)0, c, /*reuse_a=*/false, /*reuse_b=*/false);
}

// C/D tile: lane = 16*hi + n holds rows m = hi*8 + r (r = VGPR index), col n.
DEV void store_tile_bf16(unsigned short* __restrict__ dst, int row, int col,
                         int ldd, v8f c, int lane) {
  const int n = lane & 15, hi = lane >> 4;
#pragma unroll
  for (int r = 0; r < 8; ++r)
    dst[(size_t)(row + hi * 8 + r) * (size_t)ldd + (size_t)(col + n)] =
        f2bf(c[r]);
}

// Transposed store: per lane the 8 rows are consecutive -> one 16B store.
DEV void store_tile_bf16_T(unsigned short* __restrict__ dst, int row, int col,
                           int ldT, v8f c, int lane) {
  const int n = lane & 15, hi = lane >> 4;
  us8 o;
#pragma unroll
  for (int r = 0; r < 8; ++r) o[r] = f2bf(c[r]);
  *(us8*)(dst + (size_t)(col + n) * (size_t)ldT + (size_t)(row + hi * 8)) = o;
}

// Async 16-byte global -> LDS copy (ASYNCcnt-tracked Tensor/async path).
// LDS operand is the byte offset: generic LDS addresses keep the offset in
// the low 32 bits (ISA 10.2 aperture rule).
DEV void async_copy16(const unsigned short* gsrc, unsigned short* lds_dst) {
  unsigned lds_off = (unsigned)(size_t)lds_dst;
  asm volatile("global_load_async_to_lds_b128 %0, %1, off"
               :: "v"(lds_off), "v"(gsrc)
               : "memory");
}

DEV void wait_async_all() {
  asm volatile("s_wait_asynccnt 0x0" ::: "memory");
}

// ---------- kernels ----------

__global__ __launch_bounds__(256) void cvt_f32_to_bf16_x4(
    const float4* __restrict__ src, us4* __restrict__ dst, int n4) {
  int i = blockIdx.x * blockDim.x + threadIdx.x;
  if (i >= n4) return;
  float4 f = src[i];
  us4 o;
  o[0] = f2bf(f.x); o[1] = f2bf(f.y); o[2] = f2bf(f.z); o[3] = f2bf(f.w);
  dst[i] = o;
}

// Fused QKV projection: y = x @ W^T for wq/wk/wv, sharing the A fragment.
// Q,K stored per-head [b][h][s][dk]; V stored transposed [b][h][dk][s].
__global__ __launch_bounds__(256) void qkv_proj_kernel(
    const unsigned short* __restrict__ xb,
    const unsigned short* __restrict__ wqb,
    const unsigned short* __restrict__ wkb,
    const unsigned short* __restrict__ wvb,
    unsigned short* __restrict__ qb,
    unsigned short* __restrict__ kb,
    unsigned short* __restrict__ vtb) {
  const int lane = threadIdx.x & 31;
  const int wave = threadIdx.x >> 5;
  const int row  = (blockIdx.x * 8 + wave) * 16;  // [0,8192)
  const int h    = blockIdx.y;                    // head == 64-col slab
  const int col  = h * DK;

  const v8f vz = {0.f, 0.f, 0.f, 0.f, 0.f, 0.f, 0.f, 0.f};
  v8f cq[4], ck[4], cv[4];
#pragma unroll
  for (int t = 0; t < 4; ++t) { cq[t] = vz; ck[t] = vz; cv[t] = vz; }

#pragma unroll 4
  for (int k0 = 0; k0 < Dd; k0 += 32) {
    v16bf a = load_frag(xb, row, k0, Dd, lane);
#pragma unroll
    for (int t = 0; t < 4; ++t) {
      v16bf bq = load_frag(wqb, col + 16 * t, k0, Dd, lane);
      cq[t] = wmma_bf16(a, bq, cq[t]);
      v16bf bk = load_frag(wkb, col + 16 * t, k0, Dd, lane);
      ck[t] = wmma_bf16(a, bk, ck[t]);
      v16bf bv = load_frag(wvb, col + 16 * t, k0, Dd, lane);
      cv[t] = wmma_bf16(a, bv, cv[t]);
    }
  }

  const int bi = row >> 12;       // 4096 rows per batch
  const int s  = row & (Ss - 1);
  unsigned short* qh = qb  + (size_t)(bi * Hh + h) * Ss * DK;
  unsigned short* kh = kb  + (size_t)(bi * Hh + h) * Ss * DK;
  unsigned short* vh = vtb + (size_t)(bi * Hh + h) * DK * Ss;
#pragma unroll
  for (int t = 0; t < 4; ++t) {
    store_tile_bf16  (qh, s, 16 * t, DK, cq[t], lane);
    store_tile_bf16  (kh, s, 16 * t, DK, ck[t], lane);
    store_tile_bf16_T(vh, s, 16 * t, Ss, cv[t], lane);
  }
}

// Flash attention. All 8 waves of a block share one (b,h): K/V chunks are
// staged block-cooperatively into LDS with double-buffered async copies
// (ASYNCcnt), then consumed as WMMA fragments via ds_load_b128.
__global__ __launch_bounds__(256) void flash_attn_kernel(
    const unsigned short* __restrict__ qb,
    const unsigned short* __restrict__ kb,
    const unsigned short* __restrict__ vtb,
    unsigned short* __restrict__ ab) {
  // K chunk: 32 keys x 64 dk (row pad 64->72 shorts for bank spread)
  __shared__ unsigned short kst[2][32][72];
  // V^T chunk: 64 dk x 32 keys (row pad 32->40 shorts)
  __shared__ unsigned short vst[2][64][40];
  // per-wave P staging: 16 x 32 (row pad ->40)
  __shared__ unsigned short pstage[8][16][40];

  const int tid  = threadIdx.x;
  const int lane = tid & 31;
  const int wave = tid >> 5;
  const int widx = blockIdx.x * 8 + wave;  // [0, 4096)
  const int bh   = widx >> 8;              // shared by all waves in block
  const int qt   = widx & 255;
  const int bi   = bh >> 3, h = bh & 7;

  const unsigned short* qh = qb  + (size_t)bh * Ss * DK;
  const unsigned short* kh = kb  + (size_t)bh * Ss * DK;
  const unsigned short* vh = vtb + (size_t)bh * DK * Ss;

  // Block-cooperative segment assignment: 256 x 16B per chunk per matrix.
  const int krow = tid >> 3, kseg = tid & 7;  // 32 rows x 8 segs
  const int vrow = tid >> 2, vseg = tid & 3;  // 64 rows x 4 segs

  // Prologue: async-stage chunk 0 into buffer 0.
  async_copy16(kh + (size_t)krow * DK + kseg * 8, &kst[0][krow][kseg * 8]);
  async_copy16(vh + (size_t)vrow * Ss + vseg * 8, &vst[0][vrow][vseg * 8]);

  // Q fragments for this wave's 16 rows, K-dim = 64 -> two 16x32 frags.
  v16bf aq0 = load_frag(qh, qt * 16, 0,  DK, lane);
  v16bf aq1 = load_frag(qh, qt * 16, 32, DK, lane);

  const v8f vz = {0.f, 0.f, 0.f, 0.f, 0.f, 0.f, 0.f, 0.f};
  v8f acc[4];
#pragma unroll
  for (int t = 0; t < 4; ++t) acc[t] = vz;
  float mrow[8], lrow[8];
#pragma unroll
  for (int r = 0; r < 8; ++r) { mrow[r] = -3.0e38f; lrow[r] = 0.f; }

  const float SCALE = 0.125f;  // 1/sqrt(64)
  const float L2E   = 1.4426950408889634f;
  const int n = lane & 15, hi = lane >> 4;

  for (int kk = 0; kk < Ss; kk += 32) {
    const int cur = (kk >> 5) & 1;
    const int nxt = cur ^ 1;

    // Current buffer's async loads done (own wave), then barrier so every
    // wave's loads are done and everyone has finished reading buffer `nxt`
    // (read in the previous iteration) before we overwrite it.
    wait_async_all();
    __syncthreads();

    if (kk + 32 < Ss) {
      const int kn = kk + 32;
      async_copy16(kh + (size_t)(kn + krow) * DK + kseg * 8,
                   &kst[nxt][krow][kseg * 8]);
      async_copy16(vh + (size_t)vrow * Ss + kn + vseg * 8,
                   &vst[nxt][vrow][vseg * 8]);
    }

    const unsigned short* kl = &kst[cur][0][0];
    const unsigned short* vl = &vst[cur][0][0];

    // ---- S = Q K^T for 32 keys (two 16x16 tiles), K operand from LDS ----
    v8f s0 = vz, s1 = vz;
    {
      v16bf b0 = load_frag(kl, 0,  0,  72, lane);
      s0 = wmma_bf16(aq0, b0, s0);
      v16bf b1 = load_frag(kl, 0,  32, 72, lane);
      s0 = wmma_bf16(aq1, b1, s0);
      v16bf b2 = load_frag(kl, 16, 0,  72, lane);
      s1 = wmma_bf16(aq0, b2, s1);
      v16bf b3 = load_frag(kl, 16, 32, 72, lane);
      s1 = wmma_bf16(aq1, b3, s1);
    }
    s0 *= SCALE;
    s1 *= SCALE;

    // ---- online softmax: row max across the 16-lane group ----
    float cm[8];
#pragma unroll
    for (int r = 0; r < 8; ++r) cm[r] = fmaxf(s0[r], s1[r]);
#pragma unroll
    for (int msk = 1; msk < 16; msk <<= 1)
#pragma unroll
      for (int r = 0; r < 8; ++r)
        cm[r] = fmaxf(cm[r], __shfl_xor(cm[r], msk, 32));

    float mnew[8], alpha[8];
#pragma unroll
    for (int r = 0; r < 8; ++r) {
      mnew[r]  = fmaxf(mrow[r], cm[r]);
      alpha[r] = exp2f((mrow[r] - mnew[r]) * L2E);
      mrow[r]  = mnew[r];
    }

    float p0[8], p1[8], rs[8];
#pragma unroll
    for (int r = 0; r < 8; ++r) {
      p0[r] = exp2f((s0[r] - mnew[r]) * L2E);
      p1[r] = exp2f((s1[r] - mnew[r]) * L2E);
      rs[r] = p0[r] + p1[r];
    }
#pragma unroll
    for (int msk = 1; msk < 16; msk <<= 1)
#pragma unroll
      for (int r = 0; r < 8; ++r) rs[r] += __shfl_xor(rs[r], msk, 32);
#pragma unroll
    for (int r = 0; r < 8; ++r) lrow[r] = lrow[r] * alpha[r] + rs[r];
#pragma unroll
    for (int t = 0; t < 4; ++t)
#pragma unroll
      for (int r = 0; r < 8; ++r) acc[t][r] *= alpha[r];

    // ---- stage P (C layout) -> LDS -> reload as A fragment ----
#pragma unroll
    for (int r = 0; r < 8; ++r) {
      pstage[wave][hi * 8 + r][n]      = f2bf(p0[r]);
      pstage[wave][hi * 8 + r][16 + n] = f2bf(p1[r]);
    }
    asm volatile("s_wait_dscnt 0x0" ::: "memory");

    FragU pf;
    const unsigned short* lp = &pstage[wave][lane & 15][8 * hi];
    pf.h[0] = *(const us8*)lp;
    pf.h[1] = *(const us8*)(lp + 16);

    // ---- O += P @ V, V^T operand from LDS ----
#pragma unroll
    for (int t = 0; t < 4; ++t) {
      v16bf bv = load_frag(vl, 16 * t, 0, 40, lane);
      acc[t] = wmma_bf16(pf.v, bv, acc[t]);
    }
  }

  float inv[8];
#pragma unroll
  for (int r = 0; r < 8; ++r) inv[r] = 1.0f / lrow[r];
#pragma unroll
  for (int t = 0; t < 4; ++t) {
#pragma unroll
    for (int r = 0; r < 8; ++r) acc[t][r] *= inv[r];
    store_tile_bf16(ab, bi * Ss + qt * 16, h * DK + 16 * t, Dd, acc[t], lane);
  }
}

// Output projection: out = attn @ wo^T, fp32 result.
__global__ __launch_bounds__(256) void out_proj_kernel(
    const unsigned short* __restrict__ ab,
    const unsigned short* __restrict__ wob,
    float* __restrict__ out) {
  const int lane = threadIdx.x & 31;
  const int wave = threadIdx.x >> 5;
  const int row  = (blockIdx.x * 8 + wave) * 16;
  const int col  = blockIdx.y * DK;

  const v8f vz = {0.f, 0.f, 0.f, 0.f, 0.f, 0.f, 0.f, 0.f};
  v8f c[4];
#pragma unroll
  for (int t = 0; t < 4; ++t) c[t] = vz;

#pragma unroll 4
  for (int k0 = 0; k0 < Dd; k0 += 32) {
    v16bf a = load_frag(ab, row, k0, Dd, lane);
#pragma unroll
    for (int t = 0; t < 4; ++t) {
      v16bf b = load_frag(wob, col + 16 * t, k0, Dd, lane);
      c[t] = wmma_bf16(a, b, c[t]);
    }
  }

  const int n = lane & 15, hi = lane >> 4;
#pragma unroll
  for (int t = 0; t < 4; ++t)
#pragma unroll
    for (int r = 0; r < 8; ++r)
      out[(size_t)(row + hi * 8 + r) * Dd + (size_t)(col + 16 * t + n)] =
          c[t][r];
}

// ---------- host launcher ----------

extern "C" void kernel_launch(void* const* d_in, const int* in_sizes, int n_in,
                              void* d_out, int out_size, void* d_ws,
                              size_t ws_size, hipStream_t stream) {
  (void)in_sizes; (void)n_in; (void)out_size; (void)ws_size;
  const float* x  = (const float*)d_in[0];
  const float* wq = (const float*)d_in[1];
  const float* wk = (const float*)d_in[2];
  const float* wv = (const float*)d_in[3];
  const float* wo = (const float*)d_in[4];
  float* out = (float*)d_out;

  unsigned short* ws = (unsigned short*)d_ws;
  size_t off = 0;
  auto take = [&](size_t elems) {
    unsigned short* p = ws + off;
    off += elems;
    return p;
  };
  unsigned short* xb  = take((size_t)ROWS * Dd);
  unsigned short* wqb = take((size_t)Dd * Dd);
  unsigned short* wkb = take((size_t)Dd * Dd);
  unsigned short* wvb = take((size_t)Dd * Dd);
  unsigned short* wob = take((size_t)Dd * Dd);
  unsigned short* qb  = take((size_t)ROWS * Dd);
  unsigned short* kb  = take((size_t)ROWS * Dd);
  unsigned short* vtb = take((size_t)ROWS * Dd);
  unsigned short* ab  = take((size_t)ROWS * Dd);

  auto cvt = [&](const float* s, unsigned short* d, int nelem) {
    int n4 = nelem / 4;
    cvt_f32_to_bf16_x4<<<dim3((n4 + 255) / 256), dim3(256), 0, stream>>>(
        (const float4*)s, (us4*)d, n4);
  };
  cvt(x,  xb,  ROWS * Dd);
  cvt(wq, wqb, Dd * Dd);
  cvt(wk, wkb, Dd * Dd);
  cvt(wv, wvb, Dd * Dd);
  cvt(wo, wob, Dd * Dd);

  qkv_proj_kernel<<<dim3(ROWS / 128, Hh), dim3(256), 0, stream>>>(
      xb, wqb, wkb, wvb, qb, kb, vtb);

  flash_attn_kernel<<<dim3((Bb * Hh * (Ss / 16)) / 8), dim3(256), 0, stream>>>(
      qb, kb, vtb, ab);

  out_proj_kernel<<<dim3(ROWS / 128, Dd / DK), dim3(256), 0, stream>>>(
      ab, wob, out);
}
